// EquivariantAttention_73564199846149
// MI455X (gfx1250) — matrix operand order repaired and checked by reading, not compiled, not run on hardware
//
#include <hip/hip_runtime.h>
#include <hip/hip_bf16.h>

// ---------------- problem constants ----------------
constexpr int B_  = 2;
constexpr int T_  = 1024;
constexpr int R_  = 3;
constexpr int C_  = 512;
constexpr int HD_ = 16;
constexpr int H_  = 32;          // C/HD
constexpr int D_  = R_ * HD_;    // 48 (per-head dim after head split)
constexpr float EPS_ = 1e-5f;
constexpr float LOG2E = 1.4426950408889634f;

typedef __attribute__((ext_vector_type(16))) __bf16 v16bf;
typedef __attribute__((ext_vector_type(8)))  __bf16 v8bf;
typedef __attribute__((ext_vector_type(8)))  float  v8f;

#define WMMA_BF16(A, Bm, Cm) \
    __builtin_amdgcn_wmma_f32_16x16x32_bf16(false, (A), false, (Bm), (short)0, (Cm), false, false)

// Load 16 sequential bf16 (K-sequential B-matrix layout) via two 16B LDS loads.
__device__ __forceinline__ v16bf ldsB16(const __bf16* p) {
    v8bf lo = *(const v8bf*)p;
    v8bf hi = *(const v8bf*)(p + 8);
    v16bf r;
#pragma unroll
    for (int e = 0; e < 8; ++e) { r[e] = lo[e]; r[e + 8] = hi[e]; }
    return r;
}

// LDS strides (bank-conflict padding; keep 16B alignment of 16-element chunks)
constexpr int KSTR = 72;   // Klds [32 keys][72], K rows padded to 64 with zeros
constexpr int VSTR = 40;   // Vtlds [48 d][40 keys]
constexpr int PSTR = 40;   // per-wave P buffer [16 rows][40 cols]

// ====================================================================
// Kernel 1: flash attention -> o_mid [B,T,R,C] fp32
// grid (T/64, H, B), block 128 (4 waves; wave handles 16 query rows)
// ====================================================================
__global__ __launch_bounds__(128) void fa_kernel(
    const float* __restrict__ q, const float* __restrict__ k,
    const float* __restrict__ v, const float* __restrict__ bias,
    const bool* __restrict__ mask, float* __restrict__ o_mid)
{
    const int b = blockIdx.z, h = blockIdx.y;
    const int tid  = threadIdx.x;
    const int lane = tid & 31, wave = tid >> 5;
    const int g = lane >> 4;          // lane half (layout group)
    const int n = lane & 15;          // column / A-row selector
    const int qbase = blockIdx.x * 64 + wave * 16;

    __shared__ __attribute__((aligned(32))) __bf16 Klds[32 * KSTR];
    __shared__ __attribute__((aligned(32))) __bf16 Vtlds[48 * VSTR];
    __shared__ __attribute__((aligned(32))) __bf16 Plds[4][16 * PSTR];
    __shared__ float maskadd[32];

    // zero-pad K d-range [48,64) once (never overwritten)
    if (tid < 32) {
#pragma unroll
        for (int d = 48; d < 64; ++d) Klds[tid * KSTR + d] = (__bf16)0.0f;
    }

    // ---- Q tile as two A-matrices (K chunks 0..31, 32..63), bf16, pre-scaled
    const float scaling = 0.14433756729740643f;   // sqrt(16/3)/16
    v16bf qa[2];
#pragma unroll
    for (int c = 0; c < 2; ++c) {
#pragma unroll
        for (int e = 0; e < 16; ++e) {
            const int j = c * 32 + g * 16 + e;   // A-layout K index
            float val = 0.0f;
            if (j < D_) {
                const int r = j >> 4, hd = j & 15;
                val = q[(((size_t)b * T_ + (qbase + n)) * R_ + r) * C_ + h * HD_ + hd] * scaling;
            }
            qa[c][e] = (__bf16)val;
        }
    }

    // all-ones B matrix: P @ ones gives per-row sums broadcast in D-layout
    v16bf onesb;
#pragma unroll
    for (int e = 0; e < 16; ++e) onesb[e] = (__bf16)1.0f;

    v8f acc0, acc1, acc2;
    float mrow[8], lrow[8];
#pragma unroll
    for (int i = 0; i < 8; ++i) {
        acc0[i] = 0.0f; acc1[i] = 0.0f; acc2[i] = 0.0f;
        mrow[i] = -__builtin_inff(); lrow[i] = 0.0f;
    }

    // staging assignment: 96 threads own one (key, rank) pair -> 16 contig floats
    const int skey = tid & 31;
    const int sr   = tid >> 5;    // wave id doubles as rank id (waves 0..2 stage)

    for (int kbase = 0; kbase < T_; kbase += 32) {
        __syncthreads();              // previous iteration's LDS reads done
        // ---- stage K (row-major bf16) and V^T (d-major bf16)
        if (sr < R_) {
            const size_t gi = (((size_t)b * T_ + (kbase + skey)) * R_ + sr) * C_ + h * HD_;
            v8bf kb0, kb1;
#pragma unroll
            for (int e = 0; e < 8; ++e) {
                kb0[e] = (__bf16)k[gi + e];
                kb1[e] = (__bf16)k[gi + 8 + e];
                Vtlds[(sr * 16 + e) * VSTR + skey]     = (__bf16)v[gi + e];
                Vtlds[(sr * 16 + 8 + e) * VSTR + skey] = (__bf16)v[gi + 8 + e];
            }
            *(v8bf*)&Klds[skey * KSTR + sr * 16]     = kb0;
            *(v8bf*)&Klds[skey * KSTR + sr * 16 + 8] = kb1;
        }
        if (tid < 32)
            maskadd[tid] = mask[(size_t)b * T_ + kbase + tid] ? -__builtin_inff() : 0.0f;
        __syncthreads();

        // ---- bias tile (the 256MB stream): non-temporal loads, D-layout
        float bias0[8], bias1[8];
        const float* bp = bias + (((size_t)b * H_ + h) * T_ + qbase) * (size_t)T_ + kbase;
#pragma unroll
        for (int i = 0; i < 8; ++i) {
            const int row = i + g * 8;
            bias0[i] = __builtin_nontemporal_load(bp + (size_t)row * T_ + n);
            bias1[i] = __builtin_nontemporal_load(bp + (size_t)row * T_ + 16 + n);
        }

        // ---- scores: 16 queries x 32 keys (2 N-tiles x 2 K-chunks)
        v8f s0, s1;
#pragma unroll
        for (int i = 0; i < 8; ++i) { s0[i] = 0.0f; s1[i] = 0.0f; }
        {
            v16bf b00 = ldsB16(&Klds[n * KSTR + g * 16]);
            v16bf b01 = ldsB16(&Klds[n * KSTR + 32 + g * 16]);
            v16bf b10 = ldsB16(&Klds[(16 + n) * KSTR + g * 16]);
            v16bf b11 = ldsB16(&Klds[(16 + n) * KSTR + 32 + g * 16]);
            s0 = WMMA_BF16(qa[0], b00, s0);
            s0 = WMMA_BF16(qa[1], b01, s0);
            s1 = WMMA_BF16(qa[0], b10, s1);
            s1 = WMMA_BF16(qa[1], b11, s1);
        }
        const float ma0 = maskadd[n], ma1 = maskadd[16 + n];

        // ---- online softmax (rows i+g*8; max reduce over 16 lanes)
        float sc8[8];
#pragma unroll
        for (int i = 0; i < 8; ++i) {
            float a0 = s0[i] + bias0[i] + ma0;
            float a1 = s1[i] + bias1[i] + ma1;
            float cmax = fmaxf(a0, a1);
            for (int off = 1; off < 16; off <<= 1)
                cmax = fmaxf(cmax, __shfl_xor(cmax, off));
            const float mn = fmaxf(mrow[i], cmax);
            const float sc = __builtin_amdgcn_exp2f((mrow[i] - mn) * LOG2E);
            const float p0 = __builtin_amdgcn_exp2f((a0 - mn) * LOG2E);
            const float p1 = __builtin_amdgcn_exp2f((a1 - mn) * LOG2E);
            sc8[i] = sc;
            mrow[i] = mn;
            acc0[i] *= sc; acc1[i] *= sc; acc2[i] *= sc;
            // D-layout -> LDS (row-major) for A-layout reload
            Plds[wave][(i + g * 8) * PSTR + n]      = (__bf16)p0;
            Plds[wave][(i + g * 8) * PSTR + 16 + n] = (__bf16)p1;
        }
        __syncthreads();   // also orders per-wave P RAW conservatively

        // ---- P (16x32) as A-matrix
        v16bf pa;
        {
            v8bf lo = *(const v8bf*)&Plds[wave][n * PSTR + g * 16];
            v8bf hi = *(const v8bf*)&Plds[wave][n * PSTR + g * 16 + 8];
#pragma unroll
            for (int e = 0; e < 8; ++e) { pa[e] = lo[e]; pa[e + 8] = hi[e]; }
        }

        // row sums of P via WMMA against ones (replaces 32 bpermutes)
        v8f sums;
#pragma unroll
        for (int i = 0; i < 8; ++i) sums[i] = 0.0f;
        sums = WMMA_BF16(pa, onesb, sums);
#pragma unroll
        for (int i = 0; i < 8; ++i) lrow[i] = lrow[i] * sc8[i] + sums[i];

        // ---- V^T slabs as B, 3 WMMAs accumulate O
        v16bf bv0 = ldsB16(&Vtlds[(0 * 16 + n) * VSTR + g * 16]);
        v16bf bv1 = ldsB16(&Vtlds[(1 * 16 + n) * VSTR + g * 16]);
        v16bf bv2 = ldsB16(&Vtlds[(2 * 16 + n) * VSTR + g * 16]);
        acc0 = WMMA_BF16(pa, bv0, acc0);
        acc1 = WMMA_BF16(pa, bv1, acc1);
        acc2 = WMMA_BF16(pa, bv2, acc2);
    }

    // ---- epilogue: normalize by l, scatter to o_mid [B,T,R,C]
#pragma unroll
    for (int i = 0; i < 8; ++i) {
        const int trow = qbase + i + g * 8;
        const float invl = 1.0f / lrow[i];
        const size_t tb = ((size_t)b * T_ + trow) * R_;
        o_mid[(tb + 0) * C_ + h * HD_ + n] = acc0[i] * invl;
        o_mid[(tb + 1) * C_ + h * HD_ + n] = acc1[i] * invl;
        o_mid[(tb + 2) * C_ + h * HD_ + n] = acc2[i] * invl;
    }
}

// ====================================================================
// Kernel 2: per-token equivariant norm factor -> invbuf[B*T]
// grid B*T, block 32 (one wave per token)
// ====================================================================
__global__ __launch_bounds__(32) void norm_kernel(
    const float* __restrict__ o_mid, float* __restrict__ invbuf)
{
    const int idx = blockIdx.x;
    const int lane = threadIdx.x;
    const float* p = o_mid + (size_t)idx * (R_ * C_);
    float s = 0.0f;
    for (int i = lane; i < R_ * C_; i += 32) { const float x = p[i]; s += x * x; }
    for (int off = 1; off < 32; off <<= 1) s += __shfl_xor(s, off);
    if (lane == 0)
        invbuf[idx] = rsqrtf(s * (1.0f / C_) + EPS_ * EPS_);
}

// ====================================================================
// Kernel 3: out = (o * ln_w * inv) @ W^T   [6144x512] x [512x512]
// grid (6144/64, 512/64), block 128; wave: 16 rows x 64 cols
// ====================================================================
__global__ __launch_bounds__(128) void proj_kernel(
    const float* __restrict__ o_mid, const float* __restrict__ invbuf,
    const float* __restrict__ ln_w, const float* __restrict__ out_w,
    float* __restrict__ out)
{
    const int lane = threadIdx.x & 31, wave = threadIdx.x >> 5;
    const int g = lane >> 4, n = lane & 15;
    const int mbase  = blockIdx.x * 64 + wave * 16;
    const int cobase = blockIdx.y * 64;

    const int grow = mbase + n;              // this lane's A row
    const float inv = invbuf[grow / R_];

    v8f acc[4];
#pragma unroll
    for (int s = 0; s < 4; ++s)
#pragma unroll
        for (int i = 0; i < 8; ++i) acc[s][i] = 0.0f;

    for (int kb = 0; kb < C_; kb += 32) {
        // A: normed activations in A-layout (one contiguous 16-run per lane half)
        v16bf a;
#pragma unroll
        for (int half = 0; half < 2; ++half) {
            const int ci0 = kb + g * 16 + half * 8;
#pragma unroll
            for (int e = 0; e < 8; ++e) {
                const int ci = ci0 + e;
                a[half * 8 + e] = (__bf16)(o_mid[(size_t)grow * C_ + ci] * ln_w[ci] * inv);
            }
        }
        // B: W^T chunks; B[k][co] = out_w[co][k] -> contiguous row reads
#pragma unroll
        for (int s = 0; s < 4; ++s) {
            const int co = cobase + s * 16 + n;
            const float* wp = out_w + (size_t)co * C_ + kb + g * 16;
            v16bf bm;
#pragma unroll
            for (int e = 0; e < 16; ++e) bm[e] = (__bf16)wp[e];
            acc[s] = WMMA_BF16(a, bm, acc[s]);
        }
    }
#pragma unroll
    for (int s = 0; s < 4; ++s)
#pragma unroll
        for (int i = 0; i < 8; ++i)
            out[(size_t)(mbase + i + g * 8) * C_ + cobase + s * 16 + n] = acc[s][i];
}

// ====================================================================
extern "C" void kernel_launch(void* const* d_in, const int* in_sizes, int n_in,
                              void* d_out, int out_size, void* d_ws, size_t ws_size,
                              hipStream_t stream)
{
    const float* q    = (const float*)d_in[0];
    const float* k    = (const float*)d_in[1];
    const float* v    = (const float*)d_in[2];
    const float* bias = (const float*)d_in[3];
    const bool*  mask = (const bool*) d_in[4];
    const float* lnw  = (const float*)d_in[5];
    const float* outw = (const float*)d_in[6];
    float* out = (float*)d_out;

    float* o_mid  = (float*)d_ws;                                  // B*T*R*C fp32
    float* invbuf = o_mid + (size_t)B_ * T_ * R_ * C_;             // B*T fp32

    dim3 g1(T_ / 64, H_, B_);
    fa_kernel<<<g1, 128, 0, stream>>>(q, k, v, bias, mask, o_mid);

    norm_kernel<<<B_ * T_, 32, 0, stream>>>(o_mid, invbuf);

    dim3 g3((B_ * T_ * R_) / 64, C_ / 64);
    proj_kernel<<<g3, 128, 0, stream>>>(o_mid, invbuf, lnw, outw, out);
}
